// Vig_60507499266077
// MI455X (gfx1250) — matrix operand order repaired and checked by reading, not compiled, not run on hardware
//
#include <hip/hip_runtime.h>
#include <math.h>

// ---------------- types ----------------
typedef __attribute__((ext_vector_type(16))) __bf16 bf16x16;
typedef __attribute__((ext_vector_type(8)))  __bf16 bf16x8;
typedef __attribute__((ext_vector_type(8)))  float  f32x8;

union BF16x16 { bf16x16 v; bf16x8 h[2]; };

__host__ __device__ inline long cdivl(long a, long b) { return (a + b - 1) / b; }

#define NEG_INF (-1e9f)

// ---------------- conv1: direct NCHW->NHWC, fused BN+ReLU ----------------
__global__ void conv1_kernel(const float* __restrict__ img, const float* __restrict__ w,
                             const float* __restrict__ alpha, const float* __restrict__ delta,
                             float* __restrict__ out) {
  long t = (long)blockIdx.x * blockDim.x + threadIdx.x;
  const long total = 64L * 112 * 112;
  if (t >= total) return;
  int wo = (int)(t % 112); long t2 = t / 112;
  int ho = (int)(t2 % 112); int b = (int)(t2 / 112);
  float in[3][3][3];
#pragma unroll
  for (int ci = 0; ci < 3; ++ci)
#pragma unroll
    for (int kh = 0; kh < 3; ++kh)
#pragma unroll
      for (int kw = 0; kw < 3; ++kw) {
        int hi = ho * 2 - 1 + kh, wi = wo * 2 - 1 + kw;
        in[ci][kh][kw] = (hi >= 0 && hi < 224 && wi >= 0 && wi < 224)
                             ? img[(((long)b * 3 + ci) * 224 + hi) * 224 + wi] : 0.f;
      }
  float* op = out + t * 24;
  for (int co = 0; co < 24; ++co) {
    float acc = 0.f;
#pragma unroll
    for (int ci = 0; ci < 3; ++ci)
#pragma unroll
      for (int kh = 0; kh < 3; ++kh)
#pragma unroll
        for (int kw = 0; kw < 3; ++kw)
          acc += in[ci][kh][kw] * w[((co * 3 + ci) * 3 + kh) * 3 + kw];
    op[co] = fmaxf(alpha[co] * acc + delta[co], 0.f);
  }
}

// ---------------- im2col: NHWC f32 -> bf16 A[M,Kp], k = (kh*3+kw)*Cin + ci ----------------
__global__ void im2col_kernel(const float* __restrict__ in, __bf16* __restrict__ A,
                              int H, int W, int Cin, int stride, int Ho, int Wo, int Kp) {
  long t = (long)blockIdx.x * blockDim.x + threadIdx.x;
  long total = 64L * Ho * Wo * Kp;
  if (t >= total) return;
  int k = (int)(t % Kp); long m = t / Kp;
  __bf16 val = (__bf16)0.f;
  if (k < 9 * Cin) {
    int ci = k % Cin; int p = k / Cin; int kh = p / 3, kw = p % 3;
    int wo = (int)(m % Wo); long m2 = m / Wo;
    int ho = (int)(m2 % Ho); int b = (int)(m2 / Ho);
    int hi = ho * stride - 1 + kh, wi = wo * stride - 1 + kw;
    if (hi >= 0 && hi < H && wi >= 0 && wi < W)
      val = (__bf16)in[(((long)b * H + hi) * W + wi) * Cin + ci];
  }
  A[t] = val;
}

// ---------------- small prep kernels ----------------
__global__ void bn_prep_kernel(const float* bias, const float* g, const float* beta,
                               const float* m, const float* v,
                               float* alpha, float* delta, int C) {
  int c = blockIdx.x * blockDim.x + threadIdx.x;
  if (c >= C) return;
  float s = g[c] * rsqrtf(v[c] + 1e-5f);
  alpha[c] = s;
  delta[c] = (bias[c] - m[c]) * s + beta[c];
}

// conv weight OIHW f32 -> bf16 Wt[co][Kp], same k mapping as im2col
__global__ void convw_prep_kernel(const float* __restrict__ w, __bf16* __restrict__ Wt,
                                  int Cout, int Cin, int Kp) {
  long t = (long)blockIdx.x * blockDim.x + threadIdx.x;
  long total = (long)Cout * Kp;
  if (t >= total) return;
  int k = (int)(t % Kp); int co = (int)(t / Kp);
  __bf16 val = (__bf16)0.f;
  if (k < 9 * Cin) {
    int ci = k % Cin; int p = k / Cin; int kh = p / 3, kw = p % 3;
    val = (__bf16)w[(((long)co * Cin + ci) * 3 + kh) * 3 + kw];
  }
  Wt[t] = val;
}

// GAT weight [C,F] f32 -> bf16 WT [F,C]
__global__ void gatw_prep_kernel(const float* __restrict__ w, __bf16* __restrict__ WT,
                                 int C, int F) {
  long t = (long)blockIdx.x * blockDim.x + threadIdx.x;
  if (t >= (long)C * F) return;
  int c = (int)(t % C); int f = (int)(t / C);
  WT[t] = (__bf16)w[(long)c * F + f];
}

__global__ void f2bf_kernel(const float* __restrict__ in, __bf16* __restrict__ out, long n) {
  long t = (long)blockIdx.x * blockDim.x + threadIdx.x;
  if (t < n) out[t] = (__bf16)in[t];
}

// f32 [B,R,F] -> bf16 [B,F,Kp] transposed, K padded with zeros
__global__ void transpose_bt_kernel(const float* __restrict__ h, __bf16* __restrict__ hT,
                                    int R, int F, int Kp) {
  long t = (long)blockIdx.x * blockDim.x + threadIdx.x;
  long total = 64L * F * Kp;
  if (t >= total) return;
  int kp = (int)(t % Kp); long t2 = t / Kp;
  int f = (int)(t2 % F); int b = (int)(t2 / F);
  float v = (kp < R) ? h[((long)b * R + kp) * F + f] : 0.f;
  hT[t] = (__bf16)v;
}

// ---------------- WMMA bf16 GEMM: C[M,N] = A[M,K] * BT[N,K]^T, batched ----------------
// EPI: 0 none, 1 alpha*x+delta+relu, 2 alpha*x+delta, 3 elu
// Out-of-range rows/cols are CLAMPED on load (their products land only in C rows/cols
// that the guarded store discards), so the K loop is branch-free and pipelineable.
template <int EPI>
__global__ __launch_bounds__(256) void wmma_gemm_bt(
    const __bf16* __restrict__ A, const __bf16* __restrict__ BT, float* __restrict__ C,
    long sA, long sB, long sC, int M, int N, int K,
    const float* __restrict__ alpha, const float* __restrict__ delta) {
  const int bz = blockIdx.z;
  A += (long)bz * sA; BT += (long)bz * sB; C += (long)bz * sC;

  const int tid = threadIdx.x;
  const int lane = tid & 31;
  const int wave = tid >> 5;
  const int m0 = blockIdx.y * 128 + (wave & 3) * 32;  // 4 waves along M
  const int n0 = blockIdx.x * 128 + (wave >> 2) * 64; // 2 waves along N
  const int la = lane & 15;
  const int lh = lane >> 4;

  // Hoisted, clamped fragment base pointers (A: lanes 0-15 K 0..7/16..23, 16-31: 8..15/24..31;
  // B: lanes 0-15 K 0..7/16..23 of col n, lanes 16-31 K 8..15/24..31)
  const __bf16* aP[2];
#pragma unroll
  for (int mi = 0; mi < 2; ++mi) {
    int r = m0 + mi * 16 + la; if (r >= M) r = M - 1;
    aP[mi] = A + (long)r * K + lh * 8;
  }
  const __bf16* bP[4];
#pragma unroll
  for (int ni = 0; ni < 4; ++ni) {
    int c = n0 + ni * 16 + la; if (c >= N) c = N - 1;
    bP[ni] = BT + (long)c * K + lh * 8;
  }

  f32x8 acc[2][4];
#pragma unroll
  for (int mi = 0; mi < 2; ++mi)
#pragma unroll
    for (int ni = 0; ni < 4; ++ni)
#pragma unroll
      for (int v = 0; v < 8; ++v) acc[mi][ni][v] = 0.f;

  for (int k0 = 0; k0 < K; k0 += 32) {
    BF16x16 a[2], b[4];
#pragma unroll
    for (int mi = 0; mi < 2; ++mi) {
      a[mi].h[0] = *(const bf16x8*)(aP[mi] + k0);
      a[mi].h[1] = *(const bf16x8*)(aP[mi] + k0 + 16);
    }
#pragma unroll
    for (int ni = 0; ni < 4; ++ni) {
      b[ni].h[0] = *(const bf16x8*)(bP[ni] + k0);
      b[ni].h[1] = *(const bf16x8*)(bP[ni] + k0 + 16);
    }
    // prefetch next K-tile of the streaming A operand (global_prefetch_b8)
    if (k0 + 32 < K) {
      __builtin_prefetch(aP[0] + k0 + 32, 0, 1);
      __builtin_prefetch(aP[1] + k0 + 32, 0, 1);
    }
#pragma unroll
    for (int mi = 0; mi < 2; ++mi)
#pragma unroll
      for (int ni = 0; ni < 4; ++ni)
        acc[mi][ni] = __builtin_amdgcn_wmma_f32_16x16x32_bf16(
            false, a[mi].v, false, b[ni].v, (short)0, acc[mi][ni], false, false);
  }

#pragma unroll
  for (int mi = 0; mi < 2; ++mi)
#pragma unroll
    for (int ni = 0; ni < 4; ++ni)
#pragma unroll
      for (int v = 0; v < 8; ++v) {
        int r = m0 + mi * 16 + v + lh * 8;
        int c = n0 + ni * 16 + la;
        if (r < M && c < N) {
          float x = acc[mi][ni][v];
          if (EPI == 1 || EPI == 2) x = alpha[c] * x + delta[c];
          if (EPI == 1) x = fmaxf(x, 0.f);
          if (EPI == 3) x = (x > 0.f) ? x : (__expf(x) - 1.f);
          C[(long)r * N + c] = x;
        }
      }
}

// ---------------- KNN: gram -> top-9 -> dense adjacency row ----------------
__global__ void knn_adj_kernel(const float* __restrict__ G, float* __restrict__ adj, int Nn) {
  long t = (long)blockIdx.x * blockDim.x + threadIdx.x;
  if (t >= 64L * Nn) return;
  int i = (int)(t % Nn); int b = (int)(t / Nn);
  const float* gb = G + (long)b * Nn * Nn;
  const float* gi = gb + (long)i * Nn;
  float sqi = gb[(long)i * Nn + i];
  float bd[9]; int bj[9];
#pragma unroll
  for (int q = 0; q < 9; ++q) { bd[q] = 3.4e38f; bj[q] = 1 << 30; }
  for (int j = 0; j < Nn; ++j) {
    float d = sqi + gb[(long)j * Nn + j] - 2.f * gi[j];
    int q = 9;
    while (q > 0 && d < bd[q - 1]) --q; // ties keep earlier j (matches top_k stability)
    if (q < 9) {
      for (int p = 8; p > q; --p) { bd[p] = bd[p - 1]; bj[p] = bj[p - 1]; }
      bd[q] = d; bj[q] = j;
    }
  }
  float* row = adj + ((long)b * Nn + i) * Nn;
  for (int j = 0; j < Nn; ++j) row[j] = 0.f;
#pragma unroll
  for (int q = 0; q < 9; ++q) if (bj[q] < Nn) row[bj[q]] = 1.f;
}

// ---------------- GAT attention scalars ----------------
__global__ void edge_score_kernel(const float* __restrict__ h, const float* __restrict__ a,
                                  float* __restrict__ esrc, float* __restrict__ edst,
                                  long rows, int F) {
  long t = (long)blockIdx.x * blockDim.x + threadIdx.x;
  if (t >= rows) return;
  const float* hp = h + t * F;
  float s0 = 0.f, s1 = 0.f;
  for (int f = 0; f < F; ++f) { s0 += hp[f] * a[f]; s1 += hp[f] * a[F + f]; }
  esrc[t] = s0; edst[t] = s1;
}

// masked-softmax attention row -> bf16 [Kp] (padded with zeros)
__global__ void att_row_kernel(const float* __restrict__ esrc, const float* __restrict__ edst,
                               const float* __restrict__ adj, __bf16* __restrict__ att,
                               int Nn, int Kp) {
  long t = (long)blockIdx.x * blockDim.x + threadIdx.x;
  if (t >= 64L * Nn) return;
  int i = (int)(t % Nn); int b = (int)(t / Nn);
  const float* ar = adj + ((long)b * Nn + i) * Nn;
  const float* ed = edst + (long)b * Nn;
  float es = esrc[(long)b * Nn + i];
  float mx = -3.4e38f;
  for (int j = 0; j < Nn; ++j) {
    float e = es + ed[j]; e = (e > 0.f) ? e : 0.2f * e;
    e = (ar[j] > 0.f) ? e : NEG_INF;
    mx = fmaxf(mx, e);
  }
  float sum = 0.f;
  for (int j = 0; j < Nn; ++j) {
    float e = es + ed[j]; e = (e > 0.f) ? e : 0.2f * e;
    e = (ar[j] > 0.f) ? e : NEG_INF;
    sum += __expf(e - mx);
  }
  float inv = 1.f / sum;
  __bf16* out = att + ((long)b * Nn + i) * Kp;
  for (int j = 0; j < Nn; ++j) {
    float e = es + ed[j]; e = (e > 0.f) ? e : 0.2f * e;
    e = (ar[j] > 0.f) ? e : NEG_INF;
    out[j] = (__bf16)(__expf(e - mx) * inv);
  }
  for (int j = Nn; j < Kp; ++j) out[j] = (__bf16)0.f;
}

// ---------------- SAGPool: score, bitonic top-kk (stable), gather ----------------
__global__ __launch_bounds__(256) void sag_pool_kernel(
    const float* __restrict__ x, const float* __restrict__ adj, const float* __restrict__ w,
    float* out_f32, __bf16* out_bf16, float* adj_out, int Nn, int kk, int F) {
  __shared__ float s[256];
  __shared__ int id[256];
  int b = blockIdx.x;
  int tid = threadIdx.x;
  const float* xb = x + (long)b * Nn * F;
  float sc = -2.0f;
  if (tid < Nn) {
    float acc = 0.f;
    for (int f = 0; f < F; ++f) acc += xb[(long)tid * F + f] * w[f];
    sc = tanhf(acc);
  }
  s[tid] = sc;
  id[tid] = (tid < Nn) ? tid : (256 + tid);
  __syncthreads();
  // bitonic sort, "better-first": (score desc, idx asc) — matches lax.top_k order
  for (int k = 2; k <= 256; k <<= 1) {
    for (int j = k >> 1; j > 0; j >>= 1) {
      int ixj = tid ^ j;
      if (ixj > tid) {
        float s1 = s[tid], s2 = s[ixj];
        int i1 = id[tid], i2 = id[ixj];
        bool wrong = (s2 > s1) || (s2 == s1 && i2 < i1);
        if (((tid & k) == 0) == wrong) { s[tid] = s2; s[ixj] = s1; id[tid] = i2; id[ixj] = i1; }
      }
      __syncthreads();
    }
  }
  for (long t = tid; t < (long)kk * F; t += 256) {
    int r = (int)(t / F); int f = (int)(t % F);
    float val = xb[(long)id[r] * F + f] * s[r];
    if (out_f32)  out_f32[((long)b * kk + r) * F + f] = val;
    if (out_bf16) out_bf16[((long)b * kk + r) * F + f] = (__bf16)val;
  }
  if (adj_out) {
    const float* ab = adj + (long)b * Nn * Nn;
    for (long t = tid; t < (long)kk * kk; t += 256) {
      int r = (int)(t / kk); int c = (int)(t % kk);
      adj_out[((long)b * kk + r) * kk + c] = ab[(long)id[r] * Nn + id[c]];
    }
  }
}

// ================= host =================
static inline unsigned g1d(long n, int bs) { return (unsigned)((n + bs - 1) / bs); }

extern "C" void kernel_launch(void* const* d_in, const int* in_sizes, int n_in,
                              void* d_out, int out_size, void* d_ws, size_t ws_size,
                              hipStream_t stream) {
  (void)in_sizes; (void)n_in; (void)out_size; (void)ws_size;
  const float* image = (const float*)d_in[0];
  const float *cw[5], *cb[5], *gg[5], *gb[5], *gm[5], *gv[5];
  for (int l = 0; l < 5; ++l) {
    cw[l] = (const float*)d_in[1 + l * 6 + 0];
    cb[l] = (const float*)d_in[1 + l * 6 + 1];
    gg[l] = (const float*)d_in[1 + l * 6 + 2];
    gb[l] = (const float*)d_in[1 + l * 6 + 3];
    gm[l] = (const float*)d_in[1 + l * 6 + 4];
    gv[l] = (const float*)d_in[1 + l * 6 + 5];
  }
  const float* w_gat1 = (const float*)d_in[31];
  const float* a_gat1 = (const float*)d_in[32];
  const float* w_pool1 = (const float*)d_in[33];
  const float* w_gat2 = (const float*)d_in[34];
  const float* a_gat2 = (const float*)d_in[35];
  const float* w_pool2 = (const float*)d_in[36];

  char* ws = (char*)d_ws;
  constexpr size_t MBy = 1024ull * 1024ull;
  // arena (phase-reused)
  char* COL  = ws + 0;          // 96MB: im2col; later h2@0, h2T@14M, att2@24M, x3@26M, xb@48M, gram@54M, adj@66M
  char* R1   = ws + 96 * MBy;   // 80MB: act1; later h1@0, h1T@28M, att1@44M
  char* R2   = ws + 176 * MBy;  // 40MB: act2; later x1
  char* R3   = ws + 216 * MBy;  // 20MB: act3; later x2b@0, adj2@10M
  char* R4   = ws + 236 * MBy;  // 10MB: act4
  char* X0   = ws + 246 * MBy;  // 10MB: conv5 out = node feats [64,196,192] f32
  char* SM   = ws + 256 * MBy;  // 4MB smalls

  float* act1 = (float*)R1; float* act2 = (float*)R2;
  float* act3 = (float*)R3; float* act4 = (float*)R4;
  float* x0 = (float*)X0;
  __bf16* colbuf = (__bf16*)COL;

  float* alpha[5]; float* delta[5];
  for (int l = 0; l < 5; ++l) {
    alpha[l] = (float*)(SM + l * 1024);
    delta[l] = (float*)(SM + 8 * 1024 + l * 1024);
  }
  float* esrc1 = (float*)(SM + 16 * 1024);
  float* edst1 = (float*)(SM + 80 * 1024);
  float* esrc2 = (float*)(SM + 144 * 1024);
  float* edst2 = (float*)(SM + 208 * 1024);
  __bf16* W1T = (__bf16*)(SM + 272 * 1024);   // [512,192]
  __bf16* W2T = (__bf16*)(SM + 512 * 1024);   // [512,512]
  __bf16* Wt[5] = {nullptr,
                   (__bf16*)(SM + 1088 * 1024),   // [48,224]
                   (__bf16*)(SM + 1152 * 1024),   // [96,448]
                   (__bf16*)(SM + 1280 * 1024),   // [192,864]
                   (__bf16*)(SM + 1664 * 1024)};  // [192,1728]

  const int B = 64;
  // BN fold for all 5 layers
  const int Couts[5] = {24, 48, 96, 192, 192};
  for (int l = 0; l < 5; ++l)
    bn_prep_kernel<<<1, 256, 0, stream>>>(cb[l], gg[l], gb[l], gm[l], gv[l],
                                          alpha[l], delta[l], Couts[l]);
  // conv weight packing (layers 2..5)
  const int Cins[5]  = {3, 24, 48, 96, 192};
  const int Kps[5]   = {0, 224, 448, 864, 1728};
  for (int l = 1; l < 5; ++l) {
    long tot = (long)Couts[l] * Kps[l];
    convw_prep_kernel<<<g1d(tot, 256), 256, 0, stream>>>(cw[l], Wt[l], Couts[l], Cins[l], Kps[l]);
  }
  gatw_prep_kernel<<<g1d(512L * 192, 256), 256, 0, stream>>>(w_gat1, W1T, 192, 512);
  gatw_prep_kernel<<<g1d(512L * 512, 256), 256, 0, stream>>>(w_gat2, W2T, 512, 512);

  // ---- conv1 ----
  conv1_kernel<<<g1d(64L * 112 * 112, 256), 256, 0, stream>>>(image, cw[0], alpha[0], delta[0], act1);

  // ---- conv2..conv5: im2col + WMMA GEMM ----
  struct { const float* in; float* out; int H, W, Cin, s, Ho, Wo, Kp, Cout; int relu; } L[4] = {
      {act1, act2, 112, 112, 24, 2, 56, 56, 224, 48, 1},
      {act2, act3, 56, 56, 48, 2, 28, 28, 448, 96, 1},
      {act3, act4, 28, 28, 96, 2, 14, 14, 864, 192, 1},
      {act4, x0,   14, 14, 192, 1, 14, 14, 1728, 192, 0}};
  for (int l = 0; l < 4; ++l) {
    long M = (long)B * L[l].Ho * L[l].Wo;
    long tot = M * L[l].Kp;
    im2col_kernel<<<g1d(tot, 256), 256, 0, stream>>>(L[l].in, colbuf, L[l].H, L[l].W,
                                                     L[l].Cin, L[l].s, L[l].Ho, L[l].Wo, L[l].Kp);
    dim3 grid((unsigned)cdivl(L[l].Cout, 128), (unsigned)cdivl(M, 128), 1);
    if (L[l].relu)
      wmma_gemm_bt<1><<<grid, 256, 0, stream>>>(colbuf, Wt[l + 1], L[l].out, 0, 0, 0,
                                                (int)M, L[l].Cout, L[l].Kp, alpha[l + 1], delta[l + 1]);
    else
      wmma_gemm_bt<2><<<grid, 256, 0, stream>>>(colbuf, Wt[l + 1], L[l].out, 0, 0, 0,
                                                (int)M, L[l].Cout, L[l].Kp, alpha[l + 1], delta[l + 1]);
  }

  // ---- GNN buffers in reused arena ----
  __bf16* xb   = (__bf16*)(COL + 48 * MBy); // [64,196,192] bf16
  float*  gram = (float*)(COL + 54 * MBy);  // [64,196,196]
  float*  adj  = (float*)(COL + 66 * MBy);  // [64,196,196]
  float*  h1   = (float*)(R1 + 0);          // [64,196,512]
  __bf16* h1T  = (__bf16*)(R1 + 28 * MBy);  // [64,512,224]
  __bf16* att1 = (__bf16*)(R1 + 44 * MBy);  // [64,196,224]
  float*  x1   = (float*)(R2 + 0);          // [64,196,512]
  __bf16* x2b  = (__bf16*)(R3 + 0);         // [64,98,512]
  float*  adj2 = (float*)(R3 + 10 * MBy);   // [64,98,98]
  float*  h2   = (float*)(COL + 0);         // [64,98,512]
  __bf16* h2T  = (__bf16*)(COL + 14 * MBy); // [64,512,128]
  __bf16* att2 = (__bf16*)(COL + 24 * MBy); // [64,98,128]
  float*  x3   = (float*)(COL + 26 * MBy);  // [64,98,512]

  // x -> bf16
  f2bf_kernel<<<g1d(64L * 196 * 192, 256), 256, 0, stream>>>(x0, xb, 64L * 196 * 192);

  // gram = X @ X^T (WMMA, batched; BT operand == X gives X^T)
  {
    dim3 grid((unsigned)cdivl(196, 128), (unsigned)cdivl(196, 128), 64);
    wmma_gemm_bt<0><<<grid, 256, 0, stream>>>(xb, xb, gram, 196L * 192, 196L * 192, 196L * 196,
                                              196, 196, 192, nullptr, nullptr);
  }
  knn_adj_kernel<<<g1d(64L * 196, 256), 256, 0, stream>>>(gram, adj, 196);

  // h1 = x @ W1  (flat M = 64*196)
  {
    dim3 grid((unsigned)cdivl(512, 128), (unsigned)cdivl(64L * 196, 128), 1);
    wmma_gemm_bt<0><<<grid, 256, 0, stream>>>(xb, W1T, h1, 0, 0, 0, 64 * 196, 512, 192, nullptr, nullptr);
  }
  edge_score_kernel<<<g1d(64L * 196, 256), 256, 0, stream>>>(h1, a_gat1, esrc1, edst1, 64L * 196, 512);
  att_row_kernel<<<g1d(64L * 196, 256), 256, 0, stream>>>(esrc1, edst1, adj, att1, 196, 224);
  transpose_bt_kernel<<<g1d(64L * 512 * 224, 256), 256, 0, stream>>>(h1, h1T, 196, 512, 224);
  // x1 = elu(att1 @ h1)  (batched WMMA)
  {
    dim3 grid((unsigned)cdivl(512, 128), (unsigned)cdivl(196, 128), 64);
    wmma_gemm_bt<3><<<grid, 256, 0, stream>>>(att1, h1T, x1, 196L * 224, 512L * 224, 196L * 512,
                                              196, 512, 224, nullptr, nullptr);
  }
  // SAGPool 1: 196 -> 98
  sag_pool_kernel<<<64, 256, 0, stream>>>(x1, adj, w_pool1, nullptr, x2b, adj2, 196, 98, 512);

  // h2 = x2 @ W2 (flat M = 64*98)
  {
    dim3 grid((unsigned)cdivl(512, 128), (unsigned)cdivl(64L * 98, 128), 1);
    wmma_gemm_bt<0><<<grid, 256, 0, stream>>>(x2b, W2T, h2, 0, 0, 0, 64 * 98, 512, 512, nullptr, nullptr);
  }
  edge_score_kernel<<<g1d(64L * 98, 256), 256, 0, stream>>>(h2, a_gat2, esrc2, edst2, 64L * 98, 512);
  att_row_kernel<<<g1d(64L * 98, 256), 256, 0, stream>>>(esrc2, edst2, adj2, att2, 98, 128);
  transpose_bt_kernel<<<g1d(64L * 512 * 128, 256), 256, 0, stream>>>(h2, h2T, 98, 512, 128);
  // x3 = elu(att2 @ h2)
  {
    dim3 grid((unsigned)cdivl(512, 128), (unsigned)cdivl(98, 128), 64);
    wmma_gemm_bt<3><<<grid, 256, 0, stream>>>(att2, h2T, x3, 98L * 128, 512L * 128, 98L * 512,
                                              98, 512, 128, nullptr, nullptr);
  }
  // SAGPool 2: 98 -> 49, write final output f32
  sag_pool_kernel<<<64, 256, 0, stream>>>(x3, adj2, w_pool2, (float*)d_out, nullptr, nullptr, 98, 49, 512);
}
